// MultiHeadCrossAttention_4234837754501
// MI455X (gfx1250) — compile-verified, matrix operand order checked by reading, not run on hardware
//
#include <hip/hip_runtime.h>
#include <hip/hip_bf16.h>

typedef __attribute__((ext_vector_type(16))) __bf16 v16bf;
typedef __attribute__((ext_vector_type(8)))  float  v8f;
typedef __attribute__((ext_vector_type(4))) unsigned int tdm_g0_t;
typedef __attribute__((ext_vector_type(8))) int          tdm_g1_t;
typedef __attribute__((ext_vector_type(4))) int          tdm_g2_t;

#define BATCH  2
#define NQL    2048
#define NKL    2048
#define DMODEL 512
#define HEADS  8
#define HDIM   64
#define SCALE  0.125f   // 64^-0.5

#if defined(__has_builtin)
#if __has_builtin(__builtin_amdgcn_tensor_load_to_lds)
#define HAVE_TDM 1
#endif
#endif

union bffrag { v16bf v; unsigned w[8]; __bf16 e[16]; };
union u2bf4  { uint2 u; __bf16 b[4]; };

// K-index map for 16-bit WMMA A/B fragments (ISA 7.12.2):
// lanes 0-15 (g=0): VGPR0-3 -> K=0..7,  VGPR4-7 -> K=16..23
// lanes16-31 (g=1): VGPR0-3 -> K=8..15, VGPR4-7 -> K=24..31
__device__ __forceinline__ int kmap(int g, int i) {
  return (i < 4) ? (8 * g + 2 * i) : (16 + 8 * g + 2 * (i - 4));
}

__device__ __forceinline__ v8f wmma_bf16(v16bf a, v16bf b, v8f c) {
  return __builtin_amdgcn_wmma_f32_16x16x32_bf16(false, a, false, b, (short)0, c,
                                                 false, false);
}

// Async global->LDS copy of 16 bytes (CDNA5, tracked by ASYNCcnt).
__device__ __forceinline__ void async_b128(unsigned lds_addr,
                                           unsigned byte_off,
                                           unsigned long long sbase) {
  asm volatile("global_load_async_to_lds_b128 %0, %1, %2"
               :: "v"(lds_addr), "v"(byte_off), "s"(sbase) : "memory");
}
__device__ __forceinline__ void wait_async0() {
  asm volatile("s_wait_asynccnt 0x0" ::: "memory");
}

// ---------------------------------------------------------------------------
// Projection: out[b][h][n][d] = X[row][:] . W[h*64+d][:] + bias  (bf16 out)
// 128x64 tile, K-steps of 32. A/B tiles DMA'd to LDS as fp32 via async-to-LDS
// (double buffered); bf16 conversion happens at fragment build.
// ---------------------------------------------------------------------------
__global__ __launch_bounds__(256)
void proj_kernel(const float* __restrict__ X, const float* __restrict__ W,
                 const float* __restrict__ bias, __bf16* __restrict__ out) {
  __shared__ __align__(16) float As_f[2][128][36];  // stride 36 fl: 144B rows
  __shared__ __align__(16) float Bs_f[2][64][36];

  const int tid  = threadIdx.x;
  const int wave = tid >> 5, lane = tid & 31;
  const int g = lane >> 4, ln = lane & 15;
  const int m0 = blockIdx.x * 128;
  const int j0 = blockIdx.y * 64;

  const unsigned long long xb = (unsigned long long)(size_t)X;
  const unsigned long long wb = (unsigned long long)(size_t)W;

  v8f acc[4];
#pragma unroll
  for (int c = 0; c < 4; ++c)
#pragma unroll
    for (int r = 0; r < 8; ++r) acc[c][r] = 0.f;

  // ---- stage(kt, buf): issue async DMA for one K-step tile pair -----------
  auto stage = [&](int kt, int buf) {
    const int k0 = kt * 32;
#pragma unroll
    for (int i = 0; i < 4; ++i) {              // A: 128x32 fp32, b128 chunks
      int e = i * 256 + tid;
      int row = e >> 3, c4 = (e & 7) * 4;
      async_b128((unsigned)(size_t)&As_f[buf][row][c4],
                 (unsigned)(((m0 + row) * DMODEL + k0 + c4) * 4), xb);
    }
#pragma unroll
    for (int i = 0; i < 2; ++i) {              // B: 64x32 fp32
      int e = i * 256 + tid;
      int row = e >> 3, c4 = (e & 7) * 4;
      async_b128((unsigned)(size_t)&Bs_f[buf][row][c4],
                 (unsigned)(((j0 + row) * DMODEL + k0 + c4) * 4), wb);
    }
  };

  stage(0, 0);
  for (int kt = 0; kt < DMODEL / 32; ++kt) {
    const int buf = kt & 1;
    wait_async0();             // tile kt landed in LDS
    __syncthreads();
    if (kt + 1 < DMODEL / 32) stage(kt + 1, buf ^ 1);  // overlap with WMMAs

    bffrag a;
#pragma unroll
    for (int i = 0; i < 8; ++i) {
      float2 f = *(const float2*)&As_f[buf][wave * 16 + ln][kmap(g, i)];
      a.e[2 * i + 0] = (__bf16)f.x;
      a.e[2 * i + 1] = (__bf16)f.y;
    }
#pragma unroll
    for (int c = 0; c < 4; ++c) {
      bffrag bb;
#pragma unroll
      for (int i = 0; i < 8; ++i) {
        float2 f = *(const float2*)&Bs_f[buf][c * 16 + ln][kmap(g, i)];
        bb.e[2 * i + 0] = (__bf16)f.x;
        bb.e[2 * i + 1] = (__bf16)f.y;
      }
      acc[c] = wmma_bf16(a.v, bb.v, acc[c]);
    }
  }

  // Epilogue: +bias, convert to bf16, scatter to [B][H][N][64]
#pragma unroll
  for (int c = 0; c < 4; ++c) {
    int j = j0 + c * 16 + ln;
    float bv = bias[j];
    int hh = j >> 6, dd = j & 63;
#pragma unroll
    for (int r = 0; r < 8; ++r) {
      int grow = m0 + wave * 16 + 8 * g + r;
      int bb2 = grow >> 11, n = grow & (NQL - 1);
      out[(((size_t)bb2 * HEADS + hh) * NQL + n) * HDIM + dd] =
          (__bf16)(acc[c][r] + bv);
    }
  }
}

// ---------------------------------------------------------------------------
// diag[i] = P[i*NK + i%NK]
// ---------------------------------------------------------------------------
__global__ void diag_kernel(const float* __restrict__ P, float* __restrict__ D,
                            int n, int nk) {
  int i = blockIdx.x * 256 + threadIdx.x;
  if (i < n) D[i] = P[(size_t)i * nk + (i % nk)];
}

// ---------------------------------------------------------------------------
// Fused flash-style attention; K tiles via Tensor Data Mover (double
// buffered, padded LDS rows produced by D# pad fields), V tiles transposed
// manually, P bounced through per-wave LDS for the second WMMA.
// ---------------------------------------------------------------------------
#define KT   64
#define HSTR 68
#define KSTR 68
#define PSTR 68
#define NT   (NKL / KT)

__global__ __launch_bounds__(256)
void attn_kernel(const __bf16* __restrict__ Qh, const __bf16* __restrict__ Kh,
                 const __bf16* __restrict__ Vh, const float* __restrict__ mask,
                 const float* __restrict__ dgv, float* __restrict__ out) {
  __shared__ __align__(16) __bf16 Ks[2][KT][HSTR];    // [key][dim], 136B rows
  __shared__ __align__(16) __bf16 Vt[2][HDIM][KSTR];  // [dim][key]
  __shared__ __align__(16) __bf16 Ps[8][16][PSTR];    // per-wave P staging

  const int tid  = threadIdx.x;
  const int wave = tid >> 5, lane = tid & 31;
  const int g = lane >> 4, ln = lane & 15;
  const int bh = blockIdx.y;
  const int b = bh >> 3, h = bh & 7;
  const int q0 = blockIdx.x * 128;

#ifdef HAVE_TDM
  // D# group1 (constant): data_size=2B, pad every 32 dwords (=128B row) by
  // 2 dwords (=8B) -> LDS row stride 136B = 68 bf16; tensor 64x64, tile 64x64.
  const tdm_g1_t g1 = { (int)0x03110000, (int)(64 << 16), (int)(64 << 16),
                        (int)(64 << 16), 64, 64, 0, 0 };
  const tdm_g2_t z4 = { 0, 0, 0, 0 };
#if __clang_major__ >= 23
  const tdm_g1_t z8 = { 0, 0, 0, 0, 0, 0, 0, 0 };
#endif
#endif

  // ---- stage(kt, buf): TDM for K tile + manual transpose-stage of V tile --
  auto stage = [&](int kt, int buf) {
    const int key0 = kt * KT;
#ifdef HAVE_TDM
    if (wave == 0) {
      unsigned long long ga =
          (unsigned long long)(size_t)(Kh + ((size_t)bh * NKL + key0) * HDIM);
      tdm_g0_t g0 = { 1u, (unsigned)(size_t)&Ks[buf][0][0], (unsigned)ga,
                      (unsigned)((ga >> 32) & 0x1FFFFFFu) | 0x80000000u };
#if __clang_major__ >= 23
      __builtin_amdgcn_tensor_load_to_lds(g0, g1, z4, z4, z8, 0);
#else
      __builtin_amdgcn_tensor_load_to_lds(g0, g1, z4, z4, 0);
#endif
    }
#else
#pragma unroll
    for (int i = 0; i < 4; ++i) {
      int e = i * 256 + tid;
      int key = e >> 4, qd = (e & 15) * 4;
      *(uint2*)&Ks[buf][key][qd] =
          *(const uint2*)(Kh + ((size_t)bh * NKL + key0 + key) * HDIM + qd);
    }
#endif
#pragma unroll
    for (int i = 0; i < 4; ++i) {
      int e = i * 256 + tid;
      int key = e >> 4, qd = (e & 15) * 4;
      u2bf4 t;
      t.u = *(const uint2*)(Vh + ((size_t)bh * NKL + key0 + key) * HDIM + qd);
      Vt[buf][qd + 0][key] = t.b[0];
      Vt[buf][qd + 1][key] = t.b[1];
      Vt[buf][qd + 2][key] = t.b[2];
      Vt[buf][qd + 3][key] = t.b[3];
    }
  };

  // Q fragments: 16 rows x 64 dims -> two K=32 A-fragments, straight from HBM
  bffrag qa[2];
  {
    const __bf16* qptr = Qh + ((size_t)bh * NQL + (q0 + wave * 16 + ln)) * HDIM;
#pragma unroll
    for (int kc = 0; kc < 2; ++kc)
#pragma unroll
      for (int i = 0; i < 8; ++i)
        qa[kc].w[i] = *(const unsigned*)(qptr + kc * 32 + kmap(g, i));
  }

  v8f o[4];
  float m[8], l[8];
#pragma unroll
  for (int c = 0; c < 4; ++c)
#pragma unroll
    for (int r = 0; r < 8; ++r) o[c][r] = 0.f;
#pragma unroll
  for (int r = 0; r < 8; ++r) { m[r] = -__builtin_inff(); l[r] = 0.f; }

  const float* maskbase =
      mask + ((size_t)bh * NQL + (q0 + wave * 16 + 8 * g)) * NKL;

  stage(0, 0);
  for (int kt = 0; kt < NT; ++kt) {
    const int buf = kt & 1;
    const int key0 = kt * KT;
#ifdef HAVE_TDM
    if (wave == 0) __builtin_amdgcn_s_wait_tensorcnt(0);
#endif
    __syncthreads();                       // tile kt staged & visible
    if (kt + 1 < NT) stage(kt + 1, buf ^ 1);  // DMA next tile behind WMMAs

    // S = (Q . K^T) * scale
    v8f s[4];
#pragma unroll
    for (int c = 0; c < 4; ++c) {
#pragma unroll
      for (int r = 0; r < 8; ++r) s[c][r] = 0.f;
#pragma unroll
      for (int kc = 0; kc < 2; ++kc) {
        bffrag kb;
#pragma unroll
        for (int i = 0; i < 8; ++i)
          kb.w[i] =
              *(const unsigned*)&Ks[buf][c * 16 + ln][kc * 32 + kmap(g, i)];
        s[c] = wmma_bf16(qa[kc].v, kb.v, s[c]);
      }
#pragma unroll
      for (int r = 0; r < 8; ++r) s[c][r] *= SCALE;
    }

    // Online softmax (denominator over raw scores, per the reference)
#pragma unroll
    for (int r = 0; r < 8; ++r) {
      float t = fmaxf(fmaxf(s[0][r], s[1][r]), fmaxf(s[2][r], s[3][r]));
      t = fmaxf(t, __shfl_xor(t, 1, 32));
      t = fmaxf(t, __shfl_xor(t, 2, 32));
      t = fmaxf(t, __shfl_xor(t, 4, 32));
      t = fmaxf(t, __shfl_xor(t, 8, 32));
      float mn = fmaxf(m[r], t);
      float corr = __expf(m[r] - mn);
      m[r] = mn;
      float rs = 0.f;
#pragma unroll
      for (int c = 0; c < 4; ++c) {
        float p = __expf(s[c][r] - mn);
        s[c][r] = p;
        rs += p;
      }
      rs += __shfl_xor(rs, 1, 32);
      rs += __shfl_xor(rs, 2, 32);
      rs += __shfl_xor(rs, 4, 32);
      rs += __shfl_xor(rs, 8, 32);
      l[r] = l[r] * corr + rs;
#pragma unroll
      for (int c = 0; c < 4; ++c) o[c][r] *= corr;
    }

    // Apply mask * diag(pearson), stage P as bf16 (C-layout -> A-layout LDS)
#pragma unroll
    for (int c = 0; c < 4; ++c) {
      const int key = key0 + c * 16 + ln;
      const float dg = dgv[(size_t)bh * NKL + key];
#pragma unroll
      for (int r = 0; r < 8; ++r) {
        float mk = maskbase[(size_t)r * NKL + key];
        Ps[wave][8 * g + r][c * 16 + ln] = (__bf16)(s[c][r] * mk * dg);
      }
      if (key + KT < NKL) __builtin_prefetch(&maskbase[key + KT], 0, 1);
    }

    // O += P . V   (same-wave DS ordering makes the bounce barrier-free)
#pragma unroll
    for (int kc = 0; kc < 2; ++kc) {
      bffrag pa;
#pragma unroll
      for (int i = 0; i < 8; ++i)
        pa.w[i] = *(const unsigned*)&Ps[wave][ln][kc * 32 + kmap(g, i)];
#pragma unroll
      for (int c = 0; c < 4; ++c) {
        bffrag vb;
#pragma unroll
        for (int i = 0; i < 8; ++i) {
          int key = kc * 32 + kmap(g, i);
          vb.w[i] = *(const unsigned*)&Vt[buf][c * 16 + ln][key];
        }
        o[c] = wmma_bf16(pa.v, vb.v, o[c]);
      }
    }
  }

  // Normalize, write merged-head fp32 output (B, NQ, 512)
#pragma unroll
  for (int r = 0; r < 8; ++r) {
    int qq = q0 + wave * 16 + 8 * g + r;
    float inv = 1.f / l[r];
#pragma unroll
    for (int c = 0; c < 4; ++c)
      out[((size_t)b * NQL + qq) * DMODEL + h * HDIM + c * 16 + ln] =
          o[c][r] * inv;
  }
}

// ---------------------------------------------------------------------------
extern "C" void kernel_launch(void* const* d_in, const int* in_sizes, int n_in,
                              void* d_out, int out_size, void* d_ws,
                              size_t ws_size, hipStream_t stream) {
  const float* q       = (const float*)d_in[0];
  const float* k       = (const float*)d_in[1];
  const float* v       = (const float*)d_in[2];
  const float* mask    = (const float*)d_in[3];
  const float* pearson = (const float*)d_in[4];
  const float* Wq = (const float*)d_in[5];
  const float* bq = (const float*)d_in[6];
  const float* Wk = (const float*)d_in[7];
  const float* bk = (const float*)d_in[8];
  const float* Wv = (const float*)d_in[9];
  const float* bv = (const float*)d_in[10];

  const size_t nproj = (size_t)BATCH * HEADS * NQL * HDIM;  // 2^21 elems
  __bf16* Qh = (__bf16*)d_ws;
  __bf16* Kh = Qh + nproj;
  __bf16* Vh = Kh + nproj;
  float*  dg = (float*)(Vh + nproj);

  dim3 pgrid(BATCH * NQL / 128, DMODEL / 64);
  proj_kernel<<<pgrid, 256, 0, stream>>>(q, Wq, bq, Qh);
  proj_kernel<<<pgrid, 256, 0, stream>>>(k, Wk, bk, Kh);
  proj_kernel<<<pgrid, 256, 0, stream>>>(v, Wv, bv, Vh);

  int nd = BATCH * HEADS * NKL;
  diag_kernel<<<(nd + 255) / 256, 256, 0, stream>>>(pearson, dg, nd, NKL);

  dim3 agrid(NQL / 128, BATCH * HEADS);
  attn_kernel<<<agrid, 256, 0, stream>>>(Qh, Kh, Vh, mask, dg, (float*)d_out);
}
